// RGCNLinkPredictor_58402965291712
// MI455X (gfx1250) — matrix operand order repaired and checked by reading, not compiled, not run on hardware
//
#include <hip/hip_runtime.h>

// Problem constants (match reference setup_inputs)
#define kN  100000   // entities
#define kR  50       // relations
#define kD  256      // emb dim
#define kNB 4        // bases
#define kE  300000   // edges
#define kB  8192     // positive triples
#define kBN 32768    // negative triples

typedef __attribute__((ext_vector_type(16))) __bf16 v16bf;
typedef __attribute__((ext_vector_type(8)))  float  v8f;

__device__ __forceinline__ unsigned short f32_to_bf16(float f) {
  unsigned int u = __float_as_uint(f);
  unsigned int r = u + 0x7FFFu + ((u >> 16) & 1u);   // round-to-nearest-even
  return (unsigned short)(r >> 16);
}
__device__ __forceinline__ float bf16_to_f32(unsigned short h) {
  return __uint_as_float(((unsigned int)h) << 16);
}

// gfx1250 async memory->LDS copy (32 bytes: two b128 transfers; INST_OFFSET is
// added to BOTH the LDS and global addresses per the ISA pseudocode).
__device__ __forceinline__ void async_copy_32B(unsigned lds_off,
                                               unsigned long long gaddr) {
  asm volatile(
      "global_load_async_to_lds_b128 %0, %1, off\n\t"
      "global_load_async_to_lds_b128 %0, %1, off offset:16"
      :: "v"(lds_off), "v"(gaddr) : "memory");
}
__device__ __forceinline__ void wait_async0() {
  asm volatile("s_wait_asynccnt 0x0" ::: "memory");
}

// ---------------------------------------------------------------- utilities
__global__ void fill_zero_kernel(unsigned int* __restrict__ p, size_t n) {
  size_t i = (size_t)blockIdx.x * blockDim.x + threadIdx.x;
  if (i < n) p[i] = 0u;
}

__global__ void cvt_bf16_kernel(const float* __restrict__ in,
                                unsigned short* __restrict__ out, size_t n) {
  size_t i = (size_t)blockIdx.x * blockDim.x + threadIdx.x;
  if (i < n) out[i] = f32_to_bf16(in[i]);
}

// per-(dst,rel) incoming-edge count, seg = dst*R + rel  (matches reference)
__global__ void deg_kernel(const int* __restrict__ dst, const int* __restrict__ rel,
                           unsigned int* __restrict__ deg, int ne) {
  int e = blockIdx.x * blockDim.x + threadIdx.x;
  if (e < ne) atomicAdd(&deg[(size_t)dst[e] * kR + rel[e]], 1u);
}

// ---------------------------------------------------------------- WMMA GEMM
// C[M x Ncols] = A[M x 256](bf16 row-major) * B(bf16, blocks of 256x256 row-major,
// block index = col/256 -- basis (NB,256,256) is usable directly as B).
// Block tile 128x64, 8 waves, each wave a 32x32 patch = 2x2 WMMA fragments.
// Double-buffered LDS. A tile: async copy engine (GLOBAL_LOAD_ASYNC_TO_LDS_B128,
// ASYNCcnt-tracked) lands the next tile while WMMAs run on the current one.
// B tile: through registers (needs transpose + K-pair packing), stored so each
// fragment is two aligned ds_load_b128. Row strides padded to 40 bf16 (20 dwords):
// 20*c mod 64 distinct for c=0..15 -> bank-conflict-free, 16B alignment kept.
#define TM 128
#define TN 64
#define TK 32
#define ASTR 40
#define BSTR 40

// MODE 0: f32 output at [gr*ldc + gc]
// MODE 1: bf16 output interleaved as hb[n][o*NB + b] with b = gc>>8, o = gc&255
template <int MODE>
__global__ void __launch_bounds__(256)
gemm_bf16_kernel(const unsigned short* __restrict__ A,
                 const unsigned short* __restrict__ Bmat,
                 void* __restrict__ Cout, int M, int ldc) {
  __shared__ __align__(16) unsigned short As[2][TM][ASTR];  // 2 x 128 x 40
  __shared__ __align__(16) unsigned short Bs[2][TN][BSTR];  // 2 x  64 x 40 (col-major, pair-packed K)

  const int t    = threadIdx.x;
  const int lane = t & 31;
  const int w    = t >> 5;       // wave 0..7
  const int wm   = w & 3;        // wave row group (4)
  const int wn   = w >> 2;       // wave col group (2)
  const int row0 = blockIdx.x * TM;
  const int col0 = blockIdx.y * TN;
  const int bblk = col0 >> 8;          // which 256-wide B block (basis index)
  const int colL = col0 & 255;         // column offset within that block
  const unsigned short* Bp = Bmat + (size_t)bblk * kD * kD;

  const int hf  = lane >> 4;           // lane half (0: lanes 0-15, 1: 16-31)
  const int l15 = lane & 15;

  // tile-copy roles
  const int ar  = t >> 1;              // A row 0..127
  const int as_ = (t & 1) * 16;        // A col segment {0,16}
  const int brp = t >> 4;              // B K-row-pair 0..15 (rows 2rp, 2rp+1)
  const int bcg = (t & 15) * 4;        // B col group (4 cols)

  // OOB A rows feed only C rows >= M (never stored) -> clamp, no zero-fill needed
  const int grc = (row0 + ar < M) ? (row0 + ar) : 0;

  auto issue_a_async = [&](int buf, int k0) {
    unsigned lds_off = (unsigned)(uintptr_t)(&As[buf][ar][as_]);
    unsigned long long ga =
        (unsigned long long)(uintptr_t)(A + (size_t)grc * kD + k0 + as_);
    async_copy_32B(lds_off, ga);
  };
  auto load_b_global = [&](int k0, unsigned wreg[4]) {
    uint2 r0 = *(const uint2*)(Bp + (size_t)(k0 + 2 * brp) * kD + colL + bcg);
    uint2 r1 = *(const uint2*)(Bp + (size_t)(k0 + 2 * brp + 1) * kD + colL + bcg);
    // pack (k even, k odd) bf16 pairs per column
    wreg[0] = (r0.x & 0xFFFFu) | (r1.x << 16);
    wreg[1] = (r0.x >> 16)     | (r1.x & 0xFFFF0000u);
    wreg[2] = (r0.y & 0xFFFFu) | (r1.y << 16);
    wreg[3] = (r0.y >> 16)     | (r1.y & 0xFFFF0000u);
  };
  auto store_b_lds = [&](int buf, const unsigned wreg[4]) {
#pragma unroll
    for (int i = 0; i < 4; ++i)
      ((unsigned int*)&Bs[buf][bcg + i][0])[brp] = wreg[i];
  };

  v8f c[2][2] = {};
  unsigned wreg[4];

  issue_a_async(0, 0);
  load_b_global(0, wreg);
  store_b_lds(0, wreg);
  wait_async0();
  __syncthreads();

  const int NSTEP = kD / TK;   // 8
  for (int step = 0; step < NSTEP; ++step) {
    const int cur = step & 1;
    const bool more = (step + 1) < NSTEP;
    if (more) {
      issue_a_async(cur ^ 1, (step + 1) * TK);   // DMA lands during WMMA
      load_b_global((step + 1) * TK, wreg);
    }

    union Frag { v16bf v; uint4 q[2]; };
    Frag af[2], bg[2];
#pragma unroll
    for (int i = 0; i < 2; ++i) {
      // A 16x32 bf16 fragment: VGPRs 0-7 <- K = hf*16 + 0..15 (contiguous, matches B)
      const unsigned short* p = &As[cur][wm * 32 + i * 16 + l15][hf * 16];
      af[i].q[0] = *(const uint4*)p;
      af[i].q[1] = *(const uint4*)(p + 8);
    }
#pragma unroll
    for (int j = 0; j < 2; ++j) {
      // B 32x16 bf16 fragment: VGPR v <- K pair (hf*16 + 2v, +1), contiguous in Bsw
      const unsigned short* p = &Bs[cur][wn * 32 + j * 16 + l15][hf * 16];
      bg[j].q[0] = *(const uint4*)p;
      bg[j].q[1] = *(const uint4*)(p + 8);
    }
#pragma unroll
    for (int i = 0; i < 2; ++i)
#pragma unroll
      for (int j = 0; j < 2; ++j)
        c[i][j] = __builtin_amdgcn_wmma_f32_16x16x32_bf16(
            false, af[i].v, false, bg[j].v, (short)0, c[i][j], false, false);

    if (more) {
      store_b_lds(cur ^ 1, wreg);
      wait_async0();             // drain DMA into next buffer before the barrier
    }
    __syncthreads();
  }

  // store C (f32 frag layout: VGPR v -> row v + 8*half, lane%16 -> col)
#pragma unroll
  for (int i = 0; i < 2; ++i) {
#pragma unroll
    for (int j = 0; j < 2; ++j) {
      int gc = col0 + wn * 32 + j * 16 + l15;
#pragma unroll
      for (int v = 0; v < 8; ++v) {
        int gr = row0 + wm * 32 + i * 16 + v + hf * 8;
        if (gr < M) {
          float val = c[i][j][v];
          if (MODE == 0)
            ((float*)Cout)[(size_t)gr * ldc + gc] = val;
          else
            ((unsigned short*)Cout)[(size_t)gr * (kNB * kD) +
                                    (size_t)(gc & 255) * kNB + (gc >> 8)] =
                f32_to_bf16(val);
        }
      }
    }
  }
}

// ---------------------------------------------------------------- edge scatter
// agg[dst,d] += (1/deg[dst,rel]) * sum_b comp[rel,b] * hb[src][d*NB + b]
// hb is interleaved [n][d][b]: one aligned 8B uint2 per thread, fully coalesced.
__global__ void __launch_bounds__(256)
scatter_msg_kernel(const unsigned short* __restrict__ hb,
                   const int* __restrict__ src, const int* __restrict__ dst,
                   const int* __restrict__ rel, const float* __restrict__ comp,
                   const unsigned int* __restrict__ deg,
                   float* __restrict__ agg, int ne) {
  int e = blockIdx.x;
  if (e >= ne) return;
  int s = src[e], dn = dst[e], r = rel[e];
  unsigned int dg = deg[(size_t)dn * kR + r];
  float norm = 1.0f / (float)(dg > 0u ? dg : 1u);
  float c0 = comp[r * kNB + 0] * norm;
  float c1 = comp[r * kNB + 1] * norm;
  float c2 = comp[r * kNB + 2] * norm;
  float c3 = comp[r * kNB + 3] * norm;
  int d = threadIdx.x;
  uint2 pq = *(const uint2*)(hb + (size_t)s * (kNB * kD) + (size_t)d * kNB);
  float val = c0 * bf16_to_f32((unsigned short)(pq.x & 0xFFFFu)) +
              c1 * bf16_to_f32((unsigned short)(pq.x >> 16)) +
              c2 * bf16_to_f32((unsigned short)(pq.y & 0xFFFFu)) +
              c3 * bf16_to_f32((unsigned short)(pq.y >> 16));
  atomicAdd(&agg[(size_t)dn * kD + d], val);
}

// out = (agg + xroot + bias), optional relu
__global__ void combine_kernel(const float* __restrict__ agg,
                               const float* __restrict__ xroot,
                               const float* __restrict__ bias,
                               float* __restrict__ out, int relu, size_t n) {
  size_t i = (size_t)blockIdx.x * blockDim.x + threadIdx.x;
  if (i >= n) return;
  float v = agg[i] + xroot[i] + bias[i & (kD - 1)];
  if (relu) v = fmaxf(v, 0.0f);
  out[i] = v;
}

// DistMult scoring: one wave32 per triple
__global__ void __launch_bounds__(256)
score_kernel(const float* __restrict__ x, const float* __restrict__ relemb,
             const int* __restrict__ s, const int* __restrict__ r,
             const int* __restrict__ o, float* __restrict__ out, int n) {
  int wid  = blockIdx.x * (blockDim.x >> 5) + (threadIdx.x >> 5);
  int lane = threadIdx.x & 31;
  if (wid >= n) return;
  const float* xs = x + (size_t)s[wid] * kD;
  const float* xo = x + (size_t)o[wid] * kD;
  const float* re = relemb + (size_t)r[wid] * kD;
  float acc = 0.0f;
  for (int j = lane; j < kD; j += 32) acc += xs[j] * re[j] * xo[j];
#pragma unroll
  for (int off = 16; off; off >>= 1) acc += __shfl_xor(acc, off, 32);
  if (lane == 0) out[wid] = acc;
}

// ---------------------------------------------------------------- launch
extern "C" void kernel_launch(void* const* d_in, const int* in_sizes, int n_in,
                              void* d_out, int out_size, void* d_ws, size_t ws_size,
                              hipStream_t stream) {
  const int* edge_index = (const int*)d_in[0];   // (2,E): row0=src, row1=dst
  const int* edge_type  = (const int*)d_in[1];
  const int* pos_s = (const int*)d_in[2];
  const int* pos_r = (const int*)d_in[3];
  const int* pos_o = (const int*)d_in[4];
  const int* neg_s = (const int*)d_in[5];
  const int* neg_r = (const int*)d_in[6];
  const int* neg_o = (const int*)d_in[7];
  const float* entity_emb   = (const float*)d_in[8];
  const float* relation_emb = (const float*)d_in[9];
  const float* basis[2] = {(const float*)d_in[10], (const float*)d_in[14]};
  const float* comp[2]  = {(const float*)d_in[11], (const float*)d_in[15]};
  const float* root[2]  = {(const float*)d_in[12], (const float*)d_in[16]};
  const float* bias[2]  = {(const float*)d_in[13], (const float*)d_in[17]};

  const int* src = edge_index;
  const int* dst = edge_index + kE;

  // workspace carve-up (256 B aligned)
  char* ws = (char*)d_ws;
  size_t off = 0;
  auto walloc = [&](size_t bytes) -> char* {
    char* p = ws + off;
    off += (bytes + 255u) & ~(size_t)255u;
    return p;
  };
  unsigned short* hb      = (unsigned short*)walloc((size_t)kN * kNB * kD * 2); // 205 MB
  unsigned short* x_bf    = (unsigned short*)walloc((size_t)kN * kD * 2);       //  51 MB
  unsigned short* w_bf    = (unsigned short*)walloc((size_t)kNB * kD * kD * 2);
  unsigned short* root_bf = (unsigned short*)walloc((size_t)kD * kD * 2);
  float* xroot = (float*)walloc((size_t)kN * kD * 4);
  float* agg   = (float*)walloc((size_t)kN * kD * 4);
  float* x1    = (float*)walloc((size_t)kN * kD * 4);
  float* x2    = (float*)walloc((size_t)kN * kD * 4);
  unsigned int* deg = (unsigned int*)walloc((size_t)kN * kR * 4);
  (void)in_sizes; (void)n_in; (void)out_size; (void)ws_size;

  const size_t nND = (size_t)kN * kD;
  const size_t nNR = (size_t)kN * kR;

  // per-(dst,rel) degrees (graph is fixed for both layers)
  fill_zero_kernel<<<(unsigned)((nNR + 255) / 256), 256, 0, stream>>>(deg, nNR);
  deg_kernel<<<(kE + 255) / 256, 256, 0, stream>>>(dst, edge_type, deg, kE);

  const float* xin = entity_emb;
  float* xout[2] = {x1, x2};
  for (int l = 0; l < 2; ++l) {
    // bf16 conversions for WMMA inputs
    cvt_bf16_kernel<<<(unsigned)((nND + 255) / 256), 256, 0, stream>>>(xin, x_bf, nND);
    cvt_bf16_kernel<<<(unsigned)(((size_t)kNB * kD * kD + 255) / 256), 256, 0, stream>>>(
        basis[l], w_bf, (size_t)kNB * kD * kD);
    cvt_bf16_kernel<<<(unsigned)(((size_t)kD * kD + 255) / 256), 256, 0, stream>>>(
        root[l], root_bf, (size_t)kD * kD);

    // hb = x @ basis   (N x 1024, bf16 out, interleaved [n][d][b])
    dim3 g1((kN + TM - 1) / TM, (kNB * kD) / TN);
    gemm_bf16_kernel<1><<<g1, 256, 0, stream>>>(x_bf, w_bf, (void*)hb, kN, kNB * kD);
    // xroot = x @ root (N x 256, f32 out)
    dim3 g2((kN + TM - 1) / TM, kD / TN);
    gemm_bf16_kernel<0><<<g2, 256, 0, stream>>>(x_bf, root_bf, (void*)xroot, kN, kD);

    // aggregate messages
    fill_zero_kernel<<<(unsigned)((nND + 255) / 256), 256, 0, stream>>>(
        (unsigned int*)agg, nND);
    scatter_msg_kernel<<<kE, 256, 0, stream>>>(hb, src, dst, edge_type, comp[l], deg,
                                               agg, kE);
    // combine (+bias, relu on layer 0 only)
    combine_kernel<<<(unsigned)((nND + 255) / 256), 256, 0, stream>>>(
        agg, xroot, bias[l], xout[l], l == 0 ? 1 : 0, nND);
    xin = xout[l];
  }

  // scoring: pos then neg, concatenated into d_out
  score_kernel<<<(kB + 7) / 8, 256, 0, stream>>>(x2, relation_emb, pos_s, pos_r, pos_o,
                                                 (float*)d_out, kB);
  score_kernel<<<(kBN + 7) / 8, 256, 0, stream>>>(x2, relation_emb, neg_s, neg_r, neg_o,
                                                  (float*)d_out + kB, kBN);
}
